// QGAMNet_21096879357995
// MI455X (gfx1250) — matrix operand order repaired and checked by reading, not compiled, not run on hardware
//
#include <hip/hip_runtime.h>

typedef __attribute__((ext_vector_type(16))) _Float16 v16h;
typedef __attribute__((ext_vector_type(8)))  _Float16 v8h;
typedef __attribute__((ext_vector_type(8)))  float    v8f;

#define N_SAMPLES 131072
#define DD 32
#define PP 32
#define HH 64
#define TS 16   // samples per block tile

// ---------------------------------------------------------------------------
// Prep: materialize exp(log_w_main) and exp(log_w_inter) into workspace once.
// ws layout: [0 .. 2047] = exp(log_w_main) (D*H), [2048 .. 6143] = exp(log_w_inter) (P*H*2)
// ---------------------------------------------------------------------------
__global__ void qgam_prep_kernel(const float* __restrict__ log_w_main,
                                 const float* __restrict__ log_w_inter,
                                 float* __restrict__ ws) {
    int i = blockIdx.x * blockDim.x + threadIdx.x;
    if (i < DD * HH) {
        ws[i] = __expf(log_w_main[i]);
    } else if (i < DD * HH + PP * HH * 2) {
        ws[i] = __expf(log_w_inter[i - DD * HH]);
    }
}

// fast silu: one v_exp_f32 + one v_rcp_f32 (both TRANS-pipe) instead of IEEE divide
__device__ __forceinline__ float fast_silu(float t) {
    return t * __builtin_amdgcn_rcpf(1.0f + __expf(-t));
}

// ---------------------------------------------------------------------------
// Main kernel: one 16-sample tile per block (256 threads = 8 wave32s).
// Waves 0-3 handle main features d=0..31, waves 4-7 handle inter features p=0..31
// (8 features per wave). Reductions over h=64 done with v_wmma_f32_16x16x32_f16.
// ---------------------------------------------------------------------------
__global__ __launch_bounds__(256)
void qgam_main_kernel(const float* __restrict__ X,
                      const float* __restrict__ expW,       // ws from prep
                      const float* __restrict__ bias_main,
                      const float* __restrict__ v_main,
                      const float* __restrict__ bias_inter,
                      const float* __restrict__ v_inter,
                      const float* __restrict__ W_final,
                      const float* __restrict__ b_final,
                      const int*   __restrict__ pairs,
                      float* __restrict__ outg,
                      float* __restrict__ Zg)
{
    __shared__ float    sXt[TS][DD];                               // 2 KB  X tile
    __shared__ float    sWm[DD][HH];                               // 8 KB  exp(log_w_main)
    __shared__ __attribute__((aligned(16))) _Float16 sVm[DD][HH];  // 4 KB  v_main f16
    __shared__ float    sWi[PP][HH][2];                            // 16 KB exp(log_w_inter) f32
    __shared__ __attribute__((aligned(16))) _Float16 sVi[PP][HH];  // 4 KB  v_inter f16
    __shared__ float    sBm[DD];
    __shared__ float    sBi[PP][2];
    __shared__ __attribute__((aligned(16))) _Float16 sWf[HH];      // W_final f16
    __shared__ int      sPr[PP][2];
    __shared__ __attribute__((aligned(16))) _Float16 sHm[8][TS][32]; // 8 KB per-wave A staging
    __shared__ float    sZ [TS][HH];                               // 4 KB  Z tile f32
    __shared__ __attribute__((aligned(16))) _Float16 sZh[TS][HH];  // 2 KB  Z tile f16

    const int tid  = threadIdx.x;
    const int w    = tid >> 5;      // wave id (wave32)
    const int lane = tid & 31;
    const int n0   = blockIdx.x * TS;

    // ---- Phase A: fill LDS -------------------------------------------------
    {
        float*    pWm = &sWm[0][0];
        _Float16* pVm = &sVm[0][0];
        _Float16* pVi = &sVi[0][0];
        float*    pWi = &sWi[0][0][0];
        float*    pXt = &sXt[0][0];
        for (int i = tid; i < DD * HH; i += 256) {
            pWm[i] = expW[i];
            pVm[i] = (_Float16)v_main[i];
            pVi[i] = (_Float16)v_inter[i];
        }
        for (int i = tid; i < PP * HH * 2; i += 256)
            pWi[i] = expW[DD * HH + i];
        for (int i = tid; i < TS * DD; i += 256)
            pXt[i] = X[(size_t)n0 * DD + i];
        if (tid < DD) sBm[tid] = bias_main[tid];
        if (tid < PP * 2) (&sBi[0][0])[tid] = bias_inter[tid];
        if (tid < HH) sWf[tid] = (_Float16)W_final[tid];
        if (tid < PP * 2) (&sPr[0][0])[tid] = pairs[tid];
    }
    __syncthreads();

    // ---- Phase B: per-feature silu expansion + WMMA h-reduction ------------
    // sHm staging is strictly per-wave; the ISA guarantees a wave's LDS ops
    // execute in order (DScnt, in-order), so only a compiler scheduling fence
    // is needed between the staging stores and the A-fragment loads.
    const int s = lane & 15;        // sample row this lane serves
    const int g = lane >> 4;        // half-wave group
    const int offA = g ? 8 : 0;     // A-fragment K sub-offset per ISA layout

    for (int j = 0; j < 8; ++j) {
        const int f = w * 8 + j;    // feature 0..31 = main d, 32..63 = inter p
        v8f acc;
        #pragma unroll
        for (int r = 0; r < 8; ++r) acc[r] = 0.0f;

        // per-lane, per-feature invariants
        float xa = 0.0f, xb = 0.0f;
        if (f < DD) {
            xa = sXt[s][f] - sBm[f];
        } else {
            const int p = f - DD;
            xa = sXt[s][sPr[p][0]] - sBi[p][0];
            xb = sXt[s][sPr[p][1]] - sBi[p][1];
        }

        #pragma unroll
        for (int hb = 0; hb < 2; ++hb) {
            // compute 16 silu values per lane, stage as f16 row-major [sample][k]
            v8h t0, t1;
            if (f < DD) {
                const float* wrow = &sWm[f][hb * 32 + g * 16];
                #pragma unroll
                for (int i = 0; i < 16; ++i) {
                    float sl = fast_silu(xa * wrow[i]);
                    if (i < 8) t0[i] = (_Float16)sl; else t1[i - 8] = (_Float16)sl;
                }
            } else {
                const int p = f - DD;
                const float* wrow = &sWi[p][hb * 32 + g * 16][0];
                #pragma unroll
                for (int i = 0; i < 16; ++i) {
                    float sl = fast_silu(xa * wrow[2 * i] + xb * wrow[2 * i + 1]);
                    if (i < 8) t0[i] = (_Float16)sl; else t1[i - 8] = (_Float16)sl;
                }
            }
            *(v8h*)(&sHm[w][s][g * 16])     = t0;
            *(v8h*)(&sHm[w][s][g * 16 + 8]) = t1;
            __builtin_amdgcn_sched_barrier(0);   // keep stores before fragment loads

            // A fragment: 16-bit 16x32 layout (lanes 0-15: K 0-7/16-23; 16-31: K 8-15/24-31)
            const _Float16* arow = &sHm[w][s][0];
            v8h alo = *(const v8h*)(arow + offA);
            v8h ahi = *(const v8h*)(arow + 16 + offA);
            v16h A;
            #pragma unroll
            for (int i = 0; i < 8; ++i) { A[i] = alo[i]; A[8 + i] = ahi[i]; }

            // B fragment: v[f, K-range] broadcast across all 16 columns
            const _Float16* vrow = (f < DD) ? &sVm[f][0] : &sVi[f - DD][0];
            const _Float16* bp   = vrow + hb * 32 + g * 8;
            v8h blo = *(const v8h*)(bp);
            v8h bhi = *(const v8h*)(bp + 16);
            v16h B;
            #pragma unroll
            for (int i = 0; i < 8; ++i) { B[i] = blo[i]; B[8 + i] = bhi[i]; }

            acc = __builtin_amdgcn_wmma_f32_16x16x32_f16(
                false, A, false, B, (short)0, acc, false, false);
            __builtin_amdgcn_sched_barrier(0);   // keep fragment loads before next stores
        }

        // D layout: VGPR r -> M=r (lanes 0-15) / M=8+r (lanes 16-31); cols identical
        if (lane == 0 || lane == 16) {
            const int mbase = (lane == 0) ? 0 : 8;
            #pragma unroll
            for (int r = 0; r < 8; ++r) {
                float val = acc[r];
                sZ [mbase + r][f] = val;
                sZh[mbase + r][f] = (_Float16)val;
            }
        }
    }
    __syncthreads();

    // ---- Phase C: out = Z_tile @ W_final via WMMA (wave 0) -----------------
    if (w == 0) {
        v8f oc;
        #pragma unroll
        for (int r = 0; r < 8; ++r) oc[r] = 0.0f;
        #pragma unroll
        for (int hb = 0; hb < 2; ++hb) {
            const _Float16* arow = &sZh[s][0];
            v8h alo = *(const v8h*)(arow + hb * 32 + offA);
            v8h ahi = *(const v8h*)(arow + hb * 32 + 16 + offA);
            v16h A;
            #pragma unroll
            for (int i = 0; i < 8; ++i) { A[i] = alo[i]; A[8 + i] = ahi[i]; }
            const _Float16* bp = &sWf[hb * 32 + g * 8];
            v16h B;
            #pragma unroll
            for (int i = 0; i < 16; ++i) B[i] = bp[i + (i >= 8 ? 8 : 0)];
            oc = __builtin_amdgcn_wmma_f32_16x16x32_f16(
                false, A, false, B, (short)0, oc, false, false);
        }
        const float bf = b_final[0];
        if (lane == 0) {
            #pragma unroll
            for (int r = 0; r < 8; ++r) outg[n0 + r] = oc[r] + bf;
        } else if (lane == 16) {
            #pragma unroll
            for (int r = 0; r < 8; ++r) outg[n0 + 8 + r] = oc[r] + bf;
        }
    }

    // ---- Phase C2: coalesced Z writeout ------------------------------------
    for (int i = tid; i < TS * HH; i += 256) {
        const int srow = i >> 6;
        const int scol = i & 63;
        Zg[(size_t)(n0 + srow) * HH + scol] = sZ[srow][scol];
    }
}

// ---------------------------------------------------------------------------
extern "C" void kernel_launch(void* const* d_in, const int* in_sizes, int n_in,
                              void* d_out, int out_size, void* d_ws, size_t ws_size,
                              hipStream_t stream) {
    const float* X           = (const float*)d_in[0];
    const float* log_w_main  = (const float*)d_in[1];
    const float* bias_main   = (const float*)d_in[2];
    const float* v_main      = (const float*)d_in[3];
    const float* log_w_inter = (const float*)d_in[4];
    const float* bias_inter  = (const float*)d_in[5];
    const float* v_inter     = (const float*)d_in[6];
    const float* W_final     = (const float*)d_in[7];
    const float* b_final     = (const float*)d_in[8];
    const int*   pairs       = (const int*)d_in[9];

    float* outp = (float*)d_out;           // out: N floats
    float* Zp   = outp + N_SAMPLES;        // Z:   N*64 floats
    float* ws   = (float*)d_ws;            // 6144 floats of exp'd weights

    const int prep_total = DD * HH + PP * HH * 2;   // 6144
    qgam_prep_kernel<<<(prep_total + 255) / 256, 256, 0, stream>>>(
        log_w_main, log_w_inter, ws);

    qgam_main_kernel<<<N_SAMPLES / TS, 256, 0, stream>>>(
        X, ws, bias_main, v_main, bias_inter, v_inter,
        W_final, b_final, pairs, outp, Zp);
}
